// ScriptWriter_cpre_38087769981083
// MI455X (gfx1250) — matrix operand order repaired and checked by reading, not compiled, not run on hardware
//
#include <hip/hip_runtime.h>
#include <hip/hip_bf16.h>

// ---------------------------------------------------------------------------
// MI455X / gfx1250 fused transformer-encoder (3 blocks, 1 head)
// bf16 WMMA (v_wmma_f32_16x16x32_bf16) for all matmuls, f32 accumulation.
// Global->LDS staging uses CDNA5 async-to-LDS ops (ASYNCcnt) via inline asm.
// ---------------------------------------------------------------------------

typedef __bf16 bf16;
typedef __attribute__((ext_vector_type(16))) __bf16 bf16x16;
typedef __attribute__((ext_vector_type(8)))  __bf16 bf16x8;
typedef __attribute__((ext_vector_type(8)))  float  f32x8;

#define DMODEL   200
#define KPAD     224     // 7 k-steps of 32
#define NT_TILES 13      // ceil(200/16)
#define KT_STEPS 7       // ceil(200/32)
#define LSEQ     50
#define LPAD     64
#define NROWS    (4096 * 50)
#define NEGMASK  (-4294967295.0f)   // -(2^32 - 1)

__device__ __forceinline__ f32x8 wmma_bf16(bf16x16 a, bf16x16 b, f32x8 c) {
  // D = A(16x32 bf16) * B(32x16 bf16) + C(16x16 f32)
  return __builtin_amdgcn_wmma_f32_16x16x32_bf16(
      /*neg_a=*/false, a, /*neg_b=*/false, b,
      /*c_mod=*/(short)0, c, /*reuse_a=*/false, /*reuse_b=*/false);
}

// Async global->LDS copy of 8 bytes (GLOBAL_LOAD_ASYNC_TO_LDS_B64, ASYNCcnt).
// ldsOff: per-lane LDS byte address (VDST); gAddr: per-lane 64-bit global addr.
__device__ __forceinline__ void async_g2l_b64(unsigned ldsOff,
                                              unsigned long long gAddr) {
  asm volatile("global_load_async_to_lds_b64 %0, %1, off"
               :: "v"(ldsOff), "v"(gAddr) : "memory");
}
__device__ __forceinline__ void wait_asynccnt0() {
  asm volatile("s_wait_asynccnt 0x0" ::: "memory");
}

// A-matrix fragment (16x32 bf16) from a row-major bf16 buffer (LDS).
// lane<16 holds row m=lane, K = kB..kB+15;
// lane>=16 holds row m=lane-16, K = kB+16..kB+31 (contiguous, matches B).
__device__ __forceinline__ bf16x16 load_a_frag(const bf16* __restrict__ sA,
                                               int pitch, int mBase, int kBase) {
  int lane = threadIdx.x & 31;
  int m  = mBase + (lane & 15);
  int k0 = kBase + ((lane >> 4) << 4);           // +16 for upper half-wave
  const bf16* row = sA + m * pitch + k0;
  bf16x8 lo = *(const bf16x8*)(row);             // K = k0..k0+7
  bf16x8 hi = *(const bf16x8*)(row + 8);         // K = k0+8..k0+15
  bf16x16 r;
#pragma unroll
  for (int e = 0; e < 8; ++e) { r[e] = lo[e]; r[e + 8] = hi[e]; }
  return r;
}

// B-matrix fragment (32x16 bf16) from pre-shuffled weight tiles:
// tile layout = [tile][lane][16 consecutive K], one 32B contiguous load.
__device__ __forceinline__ bf16x16 load_b_frag(const bf16* __restrict__ wt, int tileIdx) {
  int lane = threadIdx.x & 31;
  return *(const bf16x16*)(wt + (size_t)(tileIdx * 32 + lane) * 16);
}

// B-matrix fragment where B = K^T: B[k][n] = Krows[n][k]; lane n reads 16
// consecutive k elements of row n => contiguous 32B LDS read (free transpose).
__device__ __forceinline__ bf16x16 load_bT_frag(const bf16* __restrict__ sK,
                                                int pitch, int nBase, int kBase) {
  int lane = threadIdx.x & 31;
  int n  = nBase + (lane & 15);
  int k0 = kBase + ((lane >> 4) << 4);           // +16 for upper half-wave
  return *(const bf16x16*)(sK + n * pitch + k0);
}

// B-matrix fragment from row-major V in LDS: B[k][n] = V[k][n] (strided).
__device__ __forceinline__ bf16x16 load_bV_frag(const bf16* __restrict__ sV,
                                                int pitch, int nBase, int kBase) {
  int lane = threadIdx.x & 31;
  int n  = nBase + (lane & 15);
  int k0 = kBase + ((lane >> 4) << 4);
  bf16x16 r;
#pragma unroll
  for (int e = 0; e < 16; ++e) r[e] = sV[(k0 + e) * pitch + n];
  return r;
}

__device__ __forceinline__ float wave_reduce_add(float v) {
#pragma unroll
  for (int off = 16; off > 0; off >>= 1) v += __shfl_xor(v, off, 32);
  return v;
}

// ---------------------------------------------------------------------------
// Weight shuffle: f32 [K=200][N=200] -> bf16 WMMA-B tiles [nt*7+kt][lane][16]
// ---------------------------------------------------------------------------
__global__ __launch_bounds__(32) void wprep_kernel(
    const float* __restrict__ W0, const float* __restrict__ W1,
    const float* __restrict__ W2, const float* __restrict__ W3,
    const float* __restrict__ W4,
    bf16* __restrict__ T0, bf16* __restrict__ T1, bf16* __restrict__ T2,
    bf16* __restrict__ T3, bf16* __restrict__ T4) {
  const float* W; bf16* T;
  switch (blockIdx.y) {
    case 0: W = W0; T = T0; break;
    case 1: W = W1; T = T1; break;
    case 2: W = W2; T = T2; break;
    case 3: W = W3; T = T3; break;
    default: W = W4; T = T4; break;
  }
  int tile = blockIdx.x;                 // 0..90
  int nt = tile / KT_STEPS, kt = tile % KT_STEPS;
  int lane = threadIdx.x;
  int n  = nt * 16 + (lane & 15);
  int kb = kt * 32 + ((lane >> 4) << 4); // B layout: K = kb + e
  bf16x16 out;
#pragma unroll
  for (int e = 0; e < 16; ++e) {
    int k = kb + e;
    float v = (k < DMODEL && n < DMODEL) ? W[k * DMODEL + n] : 0.0f;
    out[e] = (bf16)v;
  }
  *(bf16x16*)(T + (size_t)(tile * 32 + lane) * 16) = out;
}

// ---------------------------------------------------------------------------
// Embedding gather + bf16 copy + row mask (sum(x) != 0)
// one wave per row, 8 rows per 256-thread block
// ---------------------------------------------------------------------------
__global__ __launch_bounds__(256) void embed_kernel(
    const int* __restrict__ tokens, const float* __restrict__ emb,
    float* __restrict__ x, bf16* __restrict__ xb, float* __restrict__ rowmask) {
  int row  = blockIdx.x * 8 + (threadIdx.x >> 5);
  int lane = threadIdx.x & 31;
  if (row >= NROWS) return;
  long tok = (long)tokens[row];
  const float* src = emb + tok * DMODEL;
  float s = 0.0f;
  for (int c = lane; c < DMODEL; c += 32) {
    float v = src[c];
    x[(size_t)row * DMODEL + c]  = v;
    xb[(size_t)row * DMODEL + c] = (bf16)v;
    s += v;
  }
  s = wave_reduce_add(s);
  if (lane == 0) rowmask[row] = (s != 0.0f) ? 1.0f : 0.0f;
}

// ---------------------------------------------------------------------------
// GEMM: out[M x 200] = act(A[M x 200] @ W[200 x 200] + bias)
// A bf16 row-major, W pre-shuffled tiles, 64 rows per WG (8 waves).
// A is staged via async global->LDS (no VGPR bounce); pad region zeroed by VALU.
// ---------------------------------------------------------------------------
__global__ __launch_bounds__(256) void gemm_kernel(
    const bf16* __restrict__ A, const bf16* __restrict__ Wt,
    const float* __restrict__ bias,
    bf16* __restrict__ outB, float* __restrict__ outF, int doRelu) {
  __shared__ __align__(16) bf16 sA[LPAD * KPAD];   // 28 KB
  const int tid = threadIdx.x;
  const int rowBase = blockIdx.x * 64;

  // zero the K-pad columns 200..223 (6 x u64 per row), disjoint from async dst
  for (int i = tid; i < 64 * 6; i += 256) {
    int r = i / 6, j = i % 6;
    *(unsigned long long*)(&sA[r * KPAD + DMODEL + j * 4]) = 0ull;
  }
  // async copy valid 400B per row as 50 x b64 chunks
  for (int i = tid; i < 64 * 50; i += 256) {
    int r = i / 50, c = i % 50;
    unsigned lds = (unsigned)(size_t)(&sA[r * KPAD + c * 4]);
    unsigned long long ga =
        (unsigned long long)(size_t)(A + (size_t)(rowBase + r) * DMODEL + c * 4);
    async_g2l_b64(lds, ga);
  }
  wait_asynccnt0();
  __syncthreads();

  const int wave = tid >> 5;
  const int lane = tid & 31;
  for (int t = wave; t < 4 * NT_TILES; t += 8) {
    int mt = t & 3, nt = t >> 2;
    f32x8 acc = {};
#pragma unroll
    for (int kt = 0; kt < KT_STEPS; ++kt) {
      __builtin_prefetch(Wt + (size_t)((nt * KT_STEPS + kt) * 32) * 16, 0, 3);
      bf16x16 a = load_a_frag(sA, KPAD, mt * 16, kt * 32);
      bf16x16 b = load_b_frag(Wt, nt * KT_STEPS + kt);
      acc = wmma_bf16(a, b, acc);
    }
    int n = nt * 16 + (lane & 15);
    if (n < DMODEL) {
      float bv = bias[n];
      int mBase = rowBase + mt * 16 + ((lane >> 4) << 3);
#pragma unroll
      for (int p = 0; p < 8; ++p) {
        float v = acc[p] + bv;
        if (doRelu) v = fmaxf(v, 0.0f);
        size_t off = (size_t)(mBase + p) * DMODEL + n;
        if (outB) outB[off] = (bf16)v;
        if (outF) outF[off] = v;
      }
    }
  }
}

// ---------------------------------------------------------------------------
// Attention: one WG (8 waves) per sequence. S = QK^T/sqrt(d) with key mask,
// softmax, query mask, O = P V + residual(x). Dynamic LDS ~108 KB.
// Q/K/V tiles staged with async global->LDS copies.
// ---------------------------------------------------------------------------
__global__ __launch_bounds__(256) void attention_kernel(
    const bf16* __restrict__ Q, const bf16* __restrict__ K,
    const bf16* __restrict__ V, const float* __restrict__ x,
    const float* __restrict__ rowmask,
    float* __restrict__ y, bf16* __restrict__ yb) {
  extern __shared__ __align__(16) char smem[];
  bf16*  sQ = (bf16*)smem;                 // 64*224 bf16
  bf16*  sK = sQ + LPAD * KPAD;
  bf16*  sV = sK + LPAD * KPAD;
  float* sS = (float*)(sV + LPAD * KPAD);  // 64*64 f32 scores
  bf16*  sP = (bf16*)(sS + LPAD * LPAD);   // 64*64 bf16 probs
  float* sM = (float*)(sP + LPAD * LPAD);  // 64 masks

  const int b    = blockIdx.x;
  const int tid  = threadIdx.x;
  const int wave = tid >> 5;
  const int lane = tid & 31;
  const size_t base = (size_t)b * LSEQ * DMODEL;

  if (tid < LPAD) sM[tid] = (tid < LSEQ) ? rowmask[b * LSEQ + tid] : 0.0f;

  // zero pads: K-columns 200..223 of rows 0..49, and full rows 50..63
  for (int i = tid; i < LSEQ * 6; i += 256) {
    int r = i / 6, j = i % 6;
    unsigned long long* q8 = (unsigned long long*)(&sQ[r * KPAD + DMODEL + j * 4]);
    unsigned long long* k8 = (unsigned long long*)(&sK[r * KPAD + DMODEL + j * 4]);
    unsigned long long* v8 = (unsigned long long*)(&sV[r * KPAD + DMODEL + j * 4]);
    *q8 = 0ull; *k8 = 0ull; *v8 = 0ull;
  }
  for (int i = tid; i < (LPAD - LSEQ) * (KPAD / 4); i += 256) {
    int r = LSEQ + i / (KPAD / 4), j = i % (KPAD / 4);
    *(unsigned long long*)(&sQ[r * KPAD + j * 4]) = 0ull;
    *(unsigned long long*)(&sK[r * KPAD + j * 4]) = 0ull;
    *(unsigned long long*)(&sV[r * KPAD + j * 4]) = 0ull;
  }
  // async copy 50 rows x 400B for each of Q, K, V
  for (int i = tid; i < LSEQ * 50; i += 256) {
    int r = i / 50, c = i % 50;
    size_t go = base + (size_t)r * DMODEL + c * 4;
    unsigned lq = (unsigned)(size_t)(&sQ[r * KPAD + c * 4]);
    unsigned lk = (unsigned)(size_t)(&sK[r * KPAD + c * 4]);
    unsigned lv = (unsigned)(size_t)(&sV[r * KPAD + c * 4]);
    async_g2l_b64(lq, (unsigned long long)(size_t)(Q + go));
    async_g2l_b64(lk, (unsigned long long)(size_t)(K + go));
    async_g2l_b64(lv, (unsigned long long)(size_t)(V + go));
  }
  wait_asynccnt0();
  __syncthreads();

  // --- S = Q @ K^T * scale, key mask -> NEG ---
  const float scale = 0.07071067811865475f;  // 1/sqrt(200)
  for (int t = wave; t < 16; t += 8) {
    int mt = t & 3, nt = t >> 2;
    f32x8 acc = {};
#pragma unroll
    for (int kt = 0; kt < KT_STEPS; ++kt)
      acc = wmma_bf16(load_a_frag(sQ, KPAD, mt * 16, kt * 32),
                      load_bT_frag(sK, KPAD, nt * 16, kt * 32), acc);
    int n = nt * 16 + (lane & 15);
    float km = sM[n];                        // 0 for masked & padded keys
#pragma unroll
    for (int p = 0; p < 8; ++p) {
      int m = mt * 16 + p + ((lane >> 4) << 3);
      sS[m * LPAD + n] = (km != 0.0f) ? acc[p] * scale : NEGMASK;
    }
  }
  __syncthreads();

  // --- softmax per query row (threads 0..63, one row each), query mask ---
  if (tid < LPAD) {
    int m = tid;
    float mx = -3.4e38f;
    for (int j = 0; j < LPAD; ++j) mx = fmaxf(mx, sS[m * LPAD + j]);
    float sum = 0.0f;
    for (int j = 0; j < LPAD; ++j) {
      float e = __expf(sS[m * LPAD + j] - mx);
      sS[m * LPAD + j] = e;
      sum += e;
    }
    float qm = sM[m] / sum;                  // fold query mask into 1/sum
    for (int j = 0; j < LPAD; ++j)
      sP[m * LPAD + j] = (bf16)(sS[m * LPAD + j] * qm);
  }
  __syncthreads();

  // --- O = P @ V + x (residual), write f32 + bf16 ---
  for (int t = wave; t < 4 * NT_TILES; t += 8) {
    int mt = t & 3, nt = t >> 2;
    f32x8 acc = {};
#pragma unroll
    for (int kt = 0; kt < 2; ++kt)
      acc = wmma_bf16(load_a_frag(sP, LPAD, mt * 16, kt * 32),
                      load_bV_frag(sV, KPAD, nt * 16, kt * 32), acc);
    int n = nt * 16 + (lane & 15);
    if (n < DMODEL) {
#pragma unroll
      for (int p = 0; p < 8; ++p) {
        int m = mt * 16 + p + ((lane >> 4) << 3);
        if (m < LSEQ) {
          size_t off = base + (size_t)m * DMODEL + n;
          float o = acc[p] + x[off];
          y[off]  = o;
          yb[off] = (bf16)o;
        }
      }
    }
  }
}

// ---------------------------------------------------------------------------
// LayerNorm(h2)*g + b + residual -> next x (f32 + bf16) + row mask
// one wave per row
// ---------------------------------------------------------------------------
__global__ __launch_bounds__(256) void lnres_kernel(
    const float* __restrict__ h2, const float* __restrict__ resid,
    const float* __restrict__ g, const float* __restrict__ bb,
    float* __restrict__ x, bf16* __restrict__ xb, float* __restrict__ rowmask) {
  int row  = blockIdx.x * 8 + (threadIdx.x >> 5);
  int lane = threadIdx.x & 31;
  if (row >= NROWS) return;
  const float* hr = h2 + (size_t)row * DMODEL;
  float vals[7];
  float s = 0.0f;
#pragma unroll
  for (int i = 0; i < 7; ++i) {
    int c = lane + 32 * i;
    float v = (c < DMODEL) ? hr[c] : 0.0f;
    vals[i] = v; s += v;
  }
  float mu = wave_reduce_add(s) * (1.0f / DMODEL);
  float vs = 0.0f;
#pragma unroll
  for (int i = 0; i < 7; ++i) {
    int c = lane + 32 * i;
    if (c < DMODEL) { float d = vals[i] - mu; vs += d * d; }
  }
  float var = wave_reduce_add(vs) * (1.0f / DMODEL);
  float inv = rsqrtf(var + 1e-8f);
  float s2 = 0.0f;
#pragma unroll
  for (int i = 0; i < 7; ++i) {
    int c = lane + 32 * i;
    if (c < DMODEL) {
      size_t off = (size_t)row * DMODEL + c;
      float o = (vals[i] - mu) * inv * g[c] + bb[c] + resid[off];
      x[off]  = o;
      xb[off] = (bf16)o;
      s2 += o;
    }
  }
  s2 = wave_reduce_add(s2);
  if (lane == 0) rowmask[row] = (s2 != 0.0f) ? 1.0f : 0.0f;
}

// ---------------------------------------------------------------------------
extern "C" void kernel_launch(void* const* d_in, const int* in_sizes, int n_in,
                              void* d_out, int out_size, void* d_ws, size_t ws_size,
                              hipStream_t stream) {
  const int*   tokens = (const int*)d_in[0];
  const float* emb    = (const float*)d_in[1];
  const float* Wq = (const float*)d_in[2];  const float* bq = (const float*)d_in[3];
  const float* Wk = (const float*)d_in[4];  const float* bk = (const float*)d_in[5];
  const float* Wv = (const float*)d_in[6];  const float* bv = (const float*)d_in[7];
  const float* W1 = (const float*)d_in[8];  const float* b1 = (const float*)d_in[9];
  const float* W2 = (const float*)d_in[10]; const float* b2 = (const float*)d_in[11];
  const float* lng = (const float*)d_in[12]; const float* lnb = (const float*)d_in[13];

  float* xf = (float*)d_out;               // f32 activation master (164 MB)

  const size_t nBF  = (size_t)NROWS * DMODEL * sizeof(bf16);
  const size_t nF32 = (size_t)NROWS * DMODEL * sizeof(float);
  const size_t nWT  = (size_t)NT_TILES * KT_STEPS * 32 * 16 * sizeof(bf16);

  char* p = (char*)d_ws;
  auto carve = [&](size_t bytes) {
    char* r = p; p += (bytes + 255) & ~(size_t)255; return r;
  };
  bf16*  xb  = (bf16*)carve(nBF);
  bf16*  qb  = (bf16*)carve(nBF);
  bf16*  kb  = (bf16*)carve(nBF);
  bf16*  vb  = (bf16*)carve(nBF);
  bf16*  ybb = (bf16*)carve(nBF);
  bf16*  hb  = (bf16*)carve(nBF);
  float* yf  = (float*)carve(nF32);
  float* h2  = (float*)carve(nF32);
  float* rowmask = (float*)carve((size_t)NROWS * sizeof(float));
  bf16* wtq = (bf16*)carve(nWT);
  bf16* wtk = (bf16*)carve(nWT);
  bf16* wtv = (bf16*)carve(nWT);
  bf16* wt1 = (bf16*)carve(nWT);
  bf16* wt2 = (bf16*)carve(nWT);

  const int rowBlocks  = NROWS / 8;        // 25600 (wave-per-row kernels)
  const int gemmBlocks = NROWS / 64;       // 3200
  const size_t attnLds = (size_t)(3 * LPAD * KPAD) * sizeof(bf16)
                       + (size_t)(LPAD * LPAD) * sizeof(float)
                       + (size_t)(LPAD * LPAD) * sizeof(bf16)
                       + (size_t)LPAD * sizeof(float);   // ~110.8 KB

  embed_kernel<<<rowBlocks, 256, 0, stream>>>(tokens, emb, xf, xb, rowmask);

  for (int i = 0; i < 3; ++i) {
    const size_t wOff = (size_t)i * DMODEL * DMODEL;
    const size_t bOff = (size_t)i * DMODEL;
    wprep_kernel<<<dim3(NT_TILES * KT_STEPS, 5), 32, 0, stream>>>(
        Wq + wOff, Wk + wOff, Wv + wOff, W1 + wOff, W2 + wOff,
        wtq, wtk, wtv, wt1, wt2);

    gemm_kernel<<<gemmBlocks, 256, 0, stream>>>(xb, wtq, bq + bOff, qb, nullptr, 1);
    gemm_kernel<<<gemmBlocks, 256, 0, stream>>>(xb, wtk, bk + bOff, kb, nullptr, 1);
    gemm_kernel<<<gemmBlocks, 256, 0, stream>>>(xb, wtv, bv + bOff, vb, nullptr, 1);

    attention_kernel<<<4096, 256, attnLds, stream>>>(qb, kb, vb, xf, rowmask, yf, ybb);

    gemm_kernel<<<gemmBlocks, 256, 0, stream>>>(ybb, wt1, b1 + bOff, hb, nullptr, 1);
    gemm_kernel<<<gemmBlocks, 256, 0, stream>>>(hb,  wt2, b2 + bOff, nullptr, h2, 0);

    lnres_kernel<<<rowBlocks, 256, 0, stream>>>(h2, yf, lng + bOff, lnb + bOff,
                                                xf, xb, rowmask);
  }
}
